// ResNet_rml_3719441678860
// MI455X (gfx1250) — 8-bit matrix operand order repaired and checked by reading, compiled, not run on hardware
//
#include <hip/hip_runtime.h>
#include <cstdint>
#include <cstddef>

#define EPSF 1e-5f

typedef __attribute__((ext_vector_type(8))) int v8i;


__global__ void conv1_k(const float* __restrict__ x, const float* __restrict__ w,
                        float* __restrict__ out) {
  int idx = blockIdx.x * blockDim.x + threadIdx.x;
  const int total = 256 * 2 * 1024 * 32;
  if (idx >= total) return;
  int c  = idx & 31;
  int wo = (idx >> 5) & 1023;
  int ho = (idx >> 15) & 1;
  int n  = idx >> 16;
  float s = 0.f;
#pragma unroll
  for (int kh = 0; kh < 3; ++kh)
#pragma unroll
    for (int kw = 0; kw < 3; ++kw) {
      int hi = ho + kh - 1, wi = wo + kw - 1;
      if (hi >= 0 && hi < 2 && wi >= 0 && wi < 1024)
        s += x[(n * 2 + hi) * 1024 + wi] * w[c * 9 + kh * 3 + kw];
    }
  out[idx] = s;
}

__device__ __forceinline__ signed char signb(float v) {
  return (signed char)((v > 0.f) ? 1 : ((v < 0.f) ? -1 : 0));
}

__global__ void bin4_k(const float4* __restrict__ x, char4* __restrict__ o, int total4) {
  int i = blockIdx.x * blockDim.x + threadIdx.x;
  if (i >= total4) return;
  float4 v = x[i];
  char4 r;
  r.x = signb(v.x); r.y = signb(v.y); r.z = signb(v.z); r.w = signb(v.w);
  o[i] = r;
}

__global__ void add4_k(float4* __restrict__ a, const float4* __restrict__ r, int total4) {
  int i = blockIdx.x * blockDim.x + threadIdx.x;
  if (i >= total4) return;
  float4 va = a[i], vr = r[i];
  va.x += vr.x; va.y += vr.y; va.z += vr.z; va.w += vr.w;
  a[i] = va;
}

__global__ void bn_stats1_k(const float* __restrict__ x, float* __restrict__ part,
                            int C, int cLog2, int M) {
  int tc = threadIdx.x & (C - 1);
  int tr = threadIdx.x >> cLog2;
  int rStep = 256 >> cLog2;
  int rpb = (M + gridDim.x - 1) / gridDim.x;
  int r0 = blockIdx.x * rpb;
  int r1 = r0 + rpb; if (r1 > M) r1 = M;
  float s = 0.f, sq = 0.f;
  for (int r = r0 + tr; r < r1; r += rStep) {
    float v = x[(size_t)r * C + tc];
    s += v; sq += v * v;
  }
  __shared__ float sh[512];
  sh[threadIdx.x] = s; sh[256 + threadIdx.x] = sq;
  __syncthreads();
  for (int half = rStep >> 1; half >= 1; half >>= 1) {
    if (tr < half) {
      sh[threadIdx.x]       += sh[threadIdx.x + half * C];
      sh[256 + threadIdx.x] += sh[256 + threadIdx.x + half * C];
    }
    __syncthreads();
  }
  if (tr == 0) {
    part[(size_t)blockIdx.x * 2 * C + tc]     = sh[tc];
    part[(size_t)blockIdx.x * 2 * C + C + tc] = sh[256 + tc];
  }
}

__global__ void bn_stats2_k(const float* __restrict__ part, float* __restrict__ stats,
                            int C, int NB, int M) {
  int c = threadIdx.x;
  if (c >= C) return;
  float s = 0.f, sq = 0.f;
  for (int b = 0; b < NB; ++b) {
    s  += part[(size_t)b * 2 * C + c];
    sq += part[(size_t)b * 2 * C + C + c];
  }
  float m = s / (float)M;
  stats[c] = m;
  stats[C + c] = sq / (float)M - m * m;
}

__global__ void bn_apply4_k(const float4* __restrict__ x, float4* __restrict__ out,
                            const float* __restrict__ stats, const float* __restrict__ g,
                            const float* __restrict__ b, int cMask, int C, int total4,
                            int clip) {
  int i = blockIdx.x * blockDim.x + threadIdx.x;
  if (i >= total4) return;
  int c0 = (i * 4) & cMask;
  float4 v = x[i];
  float4 r;
  float* rp = (float*)&r;
  const float* vp = (const float*)&v;
#pragma unroll
  for (int j = 0; j < 4; ++j) {
    int c = c0 + j;
    float y = (vp[j] - stats[c]) * rsqrtf(stats[C + c] + EPSF) * g[c] + b[c];
    if (clip) y = fminf(1.f, fmaxf(-1.f, y));
    rp[j] = y;
  }
  out[i] = r;
}

__global__ void bn_sign4_k(const float4* __restrict__ x, char4* __restrict__ o,
                           const float* __restrict__ stats, const float* __restrict__ g,
                           const float* __restrict__ b, int cMask, int C, int total4) {
  int i = blockIdx.x * blockDim.x + threadIdx.x;
  if (i >= total4) return;
  int c0 = (i * 4) & cMask;
  float4 v = x[i];
  const float* vp = (const float*)&v;
  char4 r;
  signed char* rp = (signed char*)&r;
#pragma unroll
  for (int j = 0; j < 4; ++j) {
    int c = c0 + j;
    float y = (vp[j] - stats[c]) * rsqrtf(stats[C + c] + EPSF) * g[c] + b[c];
    rp[j] = signb(y);
  }
  o[i] = r;
}

__global__ void pack_b_k(const float* __restrict__ w, int* __restrict__ bp,
                         int Cin, int cinMask, int cinLog2, int kk, int numKc, int total) {
  int idx = blockIdx.x * blockDim.x + threadIdx.x;
  if (idx >= total) return;
  int v    = idx & 7;
  int lane = (idx >> 3) & 31;
  int kcnt = idx >> 8;
  int kc   = kcnt % numKc;
  int nT   = kcnt / numKc;
  int col  = nT * 16 + (lane & 15);
  int kbase = kc * 64 + ((v >> 1) << 4) + ((lane < 16) ? 0 : 8) + ((v & 1) << 2);
  int kmax = kk * kk * Cin;
  unsigned dw = 0;
#pragma unroll
  for (int byte = 0; byte < 4; ++byte) {
    int k = kbase + byte;
    int sc = 0;
    if (k < kmax) {
      int r  = k >> cinLog2;
      int ci = k & cinMask;
      int kh = 0, kw = 0;
      if (kk == 3) { kh = (r >= 6) ? 2 : ((r >= 3) ? 1 : 0); kw = r - kh * 3; }
      float wv = w[((size_t)(col * Cin + ci) * kk + kh) * kk + kw];
      sc = (wv > 0.f) ? 1 : ((wv < 0.f) ? -1 : 0);
    }
    dw |= ((unsigned)(sc & 0xFF)) << (byte * 8);
  }
  bp[idx] = (int)dw;
}

__device__ __forceinline__ v8i load_bfrag(const int* ldsB, int kc, int lane) {
  const int4* p = (const int4*)(ldsB + kc * 256 + lane * 8);
  int4 a = p[0], b = p[1];
  v8i r;
  r[0] = a.x; r[1] = a.y; r[2] = a.z; r[3] = a.w;
  r[4] = b.x; r[5] = b.y; r[6] = b.z; r[7] = b.w;
  return r;
}

template <int KK, int CINL, int HI>
__device__ __forceinline__ void gather_a(const signed char* const* actn,
                                         const int* hbase, const int* wbase,
                                         int kc, int laneOff, int Wi,
                                         v8i& af0, v8i& af1) {
  constexpr int Cin  = 1 << CINL;
  constexpr int kmax = KK * KK * Cin;
#pragma unroll
  for (int v = 0; v < 8; ++v) {
    const int base = kc * 64 + ((v >> 1) << 4) + ((v & 1) << 2);
    const int r  = base >> CINL;
    const int ciBase = base & (Cin - 1);
    const int kh  = (KK == 3) ? ((r >= 6) ? 2 : ((r >= 3) ? 1 : 0)) : 0;
    const int kw2 = (KK == 3) ? (r - kh * 3) : 0;
    const bool inK = base < kmax;
    const bool rowAlive = (HI != 1) || (KK == 1) || (kh == 1);
    int ci = ciBase + laneOff;
#pragma unroll
    for (int t = 0; t < 2; ++t) {
      int dw = 0;
      if (inK && rowAlive) {
        int okh, hic;
        if (HI == 1) {
          okh = 1;
          hic = 0;
        } else {
          int hi = hbase[t] + kh;
          okh = (hi >= 0) & (hi < HI);
          hic = hi < 0 ? 0 : (hi >= HI ? HI - 1 : hi);
        }
        int wi = wbase[t] + kw2;
        int okw = (wi >= 0) & (wi < Wi);
        int wic = wi < 0 ? 0 : (wi >= Wi ? Wi - 1 : wi);
        int d = *(const int*)(actn[t] + (size_t)(hic * Wi + wic) * Cin + ci);
        dw = (okh & okw) ? d : 0;
      }
      if (t == 0) af0[v] = dw; else af1[v] = dw;
    }
  }
}

template <int NUMKC, int KK, int CINL, int HI>
__global__ void __launch_bounds__(256)
bgemm_k(const signed char* __restrict__ act, const int* __restrict__ bp,
        float* __restrict__ out, int Wi, int Wo, int woLog2, int howoLog2,
        int Cout, int stride, int Mtotal) {
  constexpr int Cin = 1 << CINL;
  constexpr int pad = (KK == 3) ? 1 : 0;
  extern __shared__ int ldsB[];
  const int lane  = threadIdx.x & 31;
  const int wave  = threadIdx.x >> 5;
  const int nTile = blockIdx.y;

  const int* bsrc = bp + (size_t)nTile * (NUMKC * 256);
  {
    unsigned ldsBase = __builtin_amdgcn_groupstaticsize();
    const int segs = NUMKC * 64;
    for (int i = threadIdx.x; i < segs; i += 256) {
      unsigned ldsAddr = ldsBase + (unsigned)i * 16u;
      unsigned voff = (unsigned)i * 16u;
      asm volatile("global_load_async_to_lds_b128 %0, %1, %2"
                   :: "v"(ldsAddr), "v"(voff), "s"(bsrc)
                   : "memory");
    }
    asm volatile("s_wait_asynccnt 0" ::: "memory");
  }
  __syncthreads();

  const int mT0 = (blockIdx.x * 8 + wave) * 2;
  int hbase[2], wbase[2];
  const signed char* actn[2];
#pragma unroll
  for (int t = 0; t < 2; ++t) {
    int m = (mT0 + t) * 16 + (lane & 15);
    if (m >= Mtotal) m = Mtotal - 1;
    int nb  = m >> howoLog2;
    int rem = m & ((1 << howoLog2) - 1);
    int ho  = rem >> woLog2;
    int wo  = rem & ((1 << woLog2) - 1);
    hbase[t] = ho * stride - pad;
    wbase[t] = wo * stride - pad;
    actn[t]  = act + (size_t)nb * HI * Wi * Cin;
  }
  const int laneOff = (lane & 16) ? 8 : 0;

  v8i acc0 = {0, 0, 0, 0, 0, 0, 0, 0};
  v8i acc1 = {0, 0, 0, 0, 0, 0, 0, 0};

#pragma unroll
  for (int kc = 0; kc < NUMKC; ++kc) {
    v8i bf = load_bfrag(ldsB, kc, lane);
    v8i a0, a1;
    gather_a<KK, CINL, HI>(actn, hbase, wbase, kc, laneOff, Wi, a0, a1);
    acc0 = __builtin_amdgcn_wmma_i32_16x16x64_iu8(true, a0, true, bf, acc0, false, false);
    acc1 = __builtin_amdgcn_wmma_i32_16x16x64_iu8(true, a1, true, bf, acc1, false, false);
  }

  int col    = nTile * 16 + (lane & 15);
  int rowOff = (lane >> 4) << 3;
#pragma unroll
  for (int r = 0; r < 8; ++r) {
    int m0 = mT0 * 16 + rowOff + r;
    if (m0 < Mtotal) out[(size_t)m0 * Cout + col] = (float)acc0[r];
    int m1 = (mT0 + 1) * 16 + rowOff + r;
    if (m1 < Mtotal) out[(size_t)m1 * Cout + col] = (float)acc1[r];
  }
}

__global__ void pool_k(const float* __restrict__ h, float* __restrict__ pool) {
  int idx = blockIdx.x * blockDim.x + threadIdx.x;
  if (idx >= 256 * 128) return;
  int c = idx & 127;
  int n = idx >> 7;
  float s = 0.f;
  for (int w = 0; w < 128; ++w) s += h[((size_t)n * 128 + w) * 128 + c];
  pool[idx] = s * (1.f / 128.f);
}

__global__ void bn1d_stats_k(const float* __restrict__ pool, float* __restrict__ stats) {
  int c = threadIdx.x;
  if (c >= 128) return;
  float s = 0.f, sq = 0.f;
  for (int n = 0; n < 256; ++n) {
    float v = pool[n * 128 + c];
    s += v; sq += v * v;
  }
  float m = s / 256.f;
  stats[c] = m;
  stats[128 + c] = sq / 256.f - m * m;
}

__global__ void fc_k(const float* __restrict__ pool, const float* __restrict__ stats,
                     const float* __restrict__ g, const float* __restrict__ b,
                     const float* __restrict__ fw, const float* __restrict__ fb,
                     float* __restrict__ out) {
  int idx = blockIdx.x * blockDim.x + threadIdx.x;
  if (idx >= 256 * 24) return;
  int k = idx % 24;
  int n = idx / 24;
  float s = fb[k];
  for (int c = 0; c < 128; ++c) {
    float y = (pool[n * 128 + c] - stats[c]) * rsqrtf(stats[128 + c] + EPSF) * g[c] + b[c];
    s += y * fw[k * 128 + c];
  }
  out[n * 24 + k] = s;
}


extern "C" void kernel_launch(void* const* d_in, const int* in_sizes, int n_in,
                              void* d_out, int out_size, void* d_ws, size_t ws_size,
                              hipStream_t stream) {
  (void)n_in; (void)out_size; (void)ws_size;

  struct BlkP { const float *c1, *g1, *b1, *c2, *g2, *b2, *dw, *dg, *db; };
  const int hasDs[8] = {0, 0, 1, 0, 1, 0, 1, 0};
  const int dims[8][7] = {
      {32, 32, 1, 2, 1024, 2, 1024},
      {32, 32, 1, 2, 1024, 2, 1024},
      {32, 32, 2, 2, 1024, 1, 512},
      {32, 32, 1, 1, 512, 1, 512},
      {32, 64, 2, 1, 512, 1, 256},
      {64, 64, 1, 1, 256, 1, 256},
      {64, 128, 2, 1, 256, 1, 128},
      {128, 128, 1, 1, 128, 1, 128},
  };

  const float *x, *c1w, *bn1g, *bn1b, *bn2g, *bn2b, *fcw, *fcb;
  BlkP bk[8];
  int ii = 0;
  auto nxt = [&]() { return (const float*)d_in[ii++]; };

  if (in_sizes[0] == 524288) {
    x = nxt(); c1w = nxt(); bn1g = nxt(); bn1b = nxt();
    for (int i = 0; i < 8; ++i) {
      bk[i].c1 = nxt(); bk[i].g1 = nxt(); bk[i].b1 = nxt();
      bk[i].c2 = nxt(); bk[i].g2 = nxt(); bk[i].b2 = nxt();
      if (hasDs[i]) { bk[i].dw = nxt(); bk[i].dg = nxt(); bk[i].db = nxt(); }
      else { bk[i].dw = bk[i].dg = bk[i].db = nullptr; }
    }
    bn2g = nxt(); bn2b = nxt(); fcw = nxt(); fcb = nxt();
  } else {
    bn1b = nxt(); bn1g = nxt(); bn2b = nxt(); bn2g = nxt();
    c1w = nxt(); fcb = nxt(); fcw = nxt();
    for (int i = 0; i < 8; ++i) {
      bk[i].b1 = nxt(); bk[i].g1 = nxt(); bk[i].b2 = nxt(); bk[i].g2 = nxt();
      bk[i].c1 = nxt(); bk[i].c2 = nxt();
      if (hasDs[i]) { bk[i].db = nxt(); bk[i].dg = nxt(); bk[i].dw = nxt(); }
      else { bk[i].dw = bk[i].dg = bk[i].db = nullptr; }
    }
    x = nxt();
  }

  char* ws = (char*)d_ws;
  size_t off = 0;
  auto alloc = [&](size_t bytes) -> void* {
    void* p = ws + off;
    off = (off + bytes + 255) & ~(size_t)255;
    return p;
  };
  const size_t ACT = (size_t)256 * 2 * 1024 * 32 * sizeof(float);
  float* P[3] = {(float*)alloc(ACT), (float*)alloc(ACT), (float*)alloc(ACT)};
  float* RES = (float*)alloc((size_t)256 * 512 * 32 * sizeof(float));
  signed char* BIN = (signed char*)alloc((size_t)256 * 2 * 1024 * 32);
  float* stats = (float*)alloc(1024 * sizeof(float));
  float* part = (float*)alloc((size_t)256 * 2 * 128 * sizeof(float));
  float* pool = (float*)alloc(256 * 128 * sizeof(float));
  int* PW = (int*)alloc(256 * 1024);

  auto grid1 = [](int n) { return dim3((unsigned)((n + 255) / 256)); };
  auto ctz = [](int v) { int r = 0; while (!(v & 1)) { v >>= 1; ++r; } return r; };

  auto packW = [&](const float* w, int Cout, int Cin, int kk) {
    int numKc = (kk * kk * Cin + 63) / 64;
    int total = (Cout / 16) * numKc * 256;
    pack_b_k<<<grid1(total), 256, 0, stream>>>(w, PW, Cin, Cin - 1, ctz(Cin), kk, numKc, total);
  };
  auto gemm = [&](const signed char* act, float* outp, int Hi, int Wi, int Ho, int Wo,
                  int Cin, int Cout, int kk, int stride) {
    int numKc = (kk * kk * Cin + 63) / 64;
    int M = 256 * Ho * Wo;
    dim3 grid((unsigned)((M / 16) / 16), (unsigned)(Cout / 16));
    size_t sh = (size_t)numKc * 1024;
    int wl = ctz(Wo), hwl = ctz(Ho * Wo);
    if (kk == 3 && Cin == 32 && Hi == 2)
      bgemm_k<5, 3, 5, 2><<<grid, 256, sh, stream>>>(act, PW, outp, Wi, Wo, wl, hwl, Cout, stride, M);
    else if (kk == 3 && Cin == 32)
      bgemm_k<5, 3, 5, 1><<<grid, 256, sh, stream>>>(act, PW, outp, Wi, Wo, wl, hwl, Cout, stride, M);
    else if (kk == 3 && Cin == 64)
      bgemm_k<9, 3, 6, 1><<<grid, 256, sh, stream>>>(act, PW, outp, Wi, Wo, wl, hwl, Cout, stride, M);
    else if (kk == 3)
      bgemm_k<18, 3, 7, 1><<<grid, 256, sh, stream>>>(act, PW, outp, Wi, Wo, wl, hwl, Cout, stride, M);
    else if (Cin == 32 && Hi == 2)
      bgemm_k<1, 1, 5, 2><<<grid, 256, sh, stream>>>(act, PW, outp, Wi, Wo, wl, hwl, Cout, stride, M);
    else if (Cin == 32)
      bgemm_k<1, 1, 5, 1><<<grid, 256, sh, stream>>>(act, PW, outp, Wi, Wo, wl, hwl, Cout, stride, M);
    else
      bgemm_k<1, 1, 6, 1><<<grid, 256, sh, stream>>>(act, PW, outp, Wi, Wo, wl, hwl, Cout, stride, M);
  };
  auto bnstats = [&](const float* buf, int C, int M) {
    bn_stats1_k<<<256, 256, 0, stream>>>(buf, part, C, ctz(C), M);
    bn_stats2_k<<<1, 128, 0, stream>>>(part, stats, C, 256, M);
  };

  {
    int total = 256 * 2 * 1024 * 32;
    conv1_k<<<grid1(total), 256, 0, stream>>>(x, c1w, P[0]);
    bnstats(P[0], 32, 256 * 2 * 1024);
    bn_apply4_k<<<grid1(total / 4), 256, 0, stream>>>((const float4*)P[0], (float4*)P[0],
                                                      stats, bn1g, bn1b, 31, 32, total / 4, 0);
  }

  int pc = 0;
  for (int i = 0; i < 8; ++i) {
    const int cin = dims[i][0], cout = dims[i][1], stride = dims[i][2];
    const int Hi = dims[i][3], Wi = dims[i][4], Ho = dims[i][5], Wo = dims[i][6];
    const int Min = 256 * Hi * Wi, Mout = 256 * Ho * Wo;
    float* cur = P[pc];
    float* O1 = P[(pc + 1) % 3];
    float* O2 = P[(pc + 2) % 3];
    int totIn = Min * cin, totOut = Mout * cout;

    bin4_k<<<grid1(totIn / 4), 256, 0, stream>>>((const float4*)cur, (char4*)BIN, totIn / 4);

    packW(bk[i].c1, cout, cin, 3);
    gemm(BIN, O1, Hi, Wi, Ho, Wo, cin, cout, 3, stride);

    if (hasDs[i]) {
      packW(bk[i].dw, cout, cin, 1);
      gemm(BIN, RES, Hi, Wi, Ho, Wo, cin, cout, 1, stride);
      bnstats(RES, cout, Mout);
      bn_apply4_k<<<grid1(totOut / 4), 256, 0, stream>>>((const float4*)RES, (float4*)RES,
                                                         stats, bk[i].dg, bk[i].db,
                                                         cout - 1, cout, totOut / 4, 0);
    }

    bnstats(O1, cout, Mout);
    bn_sign4_k<<<grid1(totOut / 4), 256, 0, stream>>>((const float4*)O1, (char4*)BIN, stats,
                                                      bk[i].g1, bk[i].b1, cout - 1, cout,
                                                      totOut / 4);

    packW(bk[i].c2, cout, cout, 3);
    gemm(BIN, O2, Ho, Wo, Ho, Wo, cout, cout, 3, 1);

    add4_k<<<grid1(totOut / 4), 256, 0, stream>>>((float4*)O2,
                                                  (const float4*)(hasDs[i] ? RES : cur),
                                                  totOut / 4);
    bnstats(O2, cout, Mout);
    bn_apply4_k<<<grid1(totOut / 4), 256, 0, stream>>>((const float4*)O2, (float4*)O2, stats,
                                                       bk[i].g2, bk[i].b2, cout - 1, cout,
                                                       totOut / 4, 1);
    pc = (pc + 2) % 3;
  }

  float* h4 = P[pc];
  pool_k<<<grid1(256 * 128), 256, 0, stream>>>(h4, pool);
  bn1d_stats_k<<<1, 128, 0, stream>>>(pool, stats);
  fc_k<<<grid1(256 * 24), 256, 0, stream>>>(pool, stats, bn2g, bn2b, fcw, fcb,
                                            (float*)d_out);
}
